// ReprogrammingLayer_21096879358454
// MI455X (gfx1250) — compile-verified, matrix operand order checked by reading, not run on hardware
//
#include <hip/hip_runtime.h>

// Problem constants: B=16,P=256 -> M1=4096 rows; D_MODEL=H*DK=1024; V=1000; D_LLM=4096.

typedef __attribute__((ext_vector_type(16))) __bf16 v16bf;
typedef __attribute__((ext_vector_type(8)))  float  v8f;

union Frag32 { uint4 u[2]; v16bf v; };

static __device__ __forceinline__ unsigned short f2bf_u(float f) {
  unsigned int u = __float_as_uint(f);
  u += 0x7FFFu + ((u >> 16) & 1u);          // round-to-nearest-even
  return (unsigned short)(u >> 16);
}
static __device__ __forceinline__ __bf16 f2bf(float f) {
  unsigned short h = f2bf_u(f);
  return __builtin_bit_cast(__bf16, h);
}

// Async global -> LDS copy of 16 bytes (gfx1250 GLOBAL_LOAD_ASYNC_TO_LDS_B128,
// GVS addressing: mem = SADDR64 + VOFF32; tracked by ASYNCcnt, no VGPR data).
static __device__ __forceinline__ void async_g2l_b128(
    unsigned lds_byte_addr, unsigned goff_bytes, unsigned long long gbase) {
  asm volatile("global_load_async_to_lds_b128 %0, %1, %2"
               :: "v"(lds_byte_addr), "v"(goff_bytes), "s"(gbase)
               : "memory");
}
static __device__ __forceinline__ void wait_async0() {
  asm volatile("s_wait_asynccnt 0x0" ::: "memory");
}

// ---- Kernel 1: column sums over V of source/value embeddings ---------------
__global__ __launch_bounds__(256) void colsum_kernel(
    const float* __restrict__ se, const float* __restrict__ ve,
    float* __restrict__ ssum, float* __restrict__ vsum) {
  const int c = blockIdx.x * 256 + threadIdx.x;          // 0..4095
  const float* p = (blockIdx.y == 0) ? se : ve;
  float acc = 0.f;
  for (int v = 0; v < 1000; ++v) acc += p[(size_t)v * 4096 + c];
  ((blockIdx.y == 0) ? ssum : vsum)[c] = acc;
}

// ---- Kernel 2: Ksum = ssum@Wk + V*bk (pre-scaled by 1/sqrt(64)),
//                Vsum = vsum@Wv + V*bv ---------------------------------------
__global__ __launch_bounds__(256) void headsum_kernel(
    const float* __restrict__ ssum, const float* __restrict__ vsum,
    const float* __restrict__ Wk, const float* __restrict__ bk,
    const float* __restrict__ Wv, const float* __restrict__ bv,
    float* __restrict__ ksc, float* __restrict__ vsh) {
  const int j = blockIdx.x * 256 + threadIdx.x;          // 0..1023
  const bool kq = (blockIdx.y == 0);
  const float* s = kq ? ssum : vsum;
  const float* W = kq ? Wk : Wv;
  const float* b = kq ? bk : bv;
  float acc = 0.f;
  for (int i = 0; i < 4096; ++i) acc += s[i] * W[(size_t)i * 1024 + j];
  acc += 1000.0f * b[j];
  if (kq) ksc[j] = acc * 0.125f;                         // fold softmax scale
  else    vsh[j] = acc;
}

// ---- Kernel 3: f32 -> bf16 (row-major copy, for the activation) ------------
__global__ __launch_bounds__(256) void cvt_bf16_kernel(
    const float* __restrict__ in, __bf16* __restrict__ out, int n) {
  int i = (blockIdx.x * 256 + threadIdx.x) * 4;
  if (i + 4 > n) return;                                 // n is multiple of 4
  float4 f = *(const float4*)(in + i);
  ushort4 o;
  o.x = f2bf_u(f.x); o.y = f2bf_u(f.y); o.z = f2bf_u(f.z); o.w = f2bf_u(f.w);
  *(ushort4*)((unsigned short*)out + i) = o;
}

// ---- Kernel 4: f32 [K][N] -> bf16 transposed [N][K] (tiled via LDS) --------
__global__ __launch_bounds__(256) void cvt_bf16_t_kernel(
    const float* __restrict__ in, __bf16* __restrict__ out, int N, int K) {
  __shared__ unsigned short t[32][36];                   // padded: no conflicts
  const int n0 = blockIdx.x * 32, k0 = blockIdx.y * 32;
  const int r = threadIdx.x >> 3;                        // 0..31
  const int c = (threadIdx.x & 7) * 4;                   // 0,4,...,28
  float4 f = *(const float4*)(in + (size_t)(k0 + r) * N + n0 + c);
  t[r][c + 0] = f2bf_u(f.x); t[r][c + 1] = f2bf_u(f.y);
  t[r][c + 2] = f2bf_u(f.z); t[r][c + 3] = f2bf_u(f.w);
  __syncthreads();
  ushort4 o;                                             // out[n0+r][k0+c..c+3]
  o.x = t[c + 0][r]; o.y = t[c + 1][r]; o.z = t[c + 2][r]; o.w = t[c + 3][r];
  *(ushort4*)((unsigned short*)out + (size_t)(n0 + r) * K + k0 + c) = o;
}

// ---- WMMA GEMM tiling ------------------------------------------------------
#define BM   256      // rows per block (8 waves x 32)
#define BN   64       // cols per block (== one head for GEMM1)
#define BK   32       // K step == WMMA K
#define KD   1024     // shared K dim of both GEMMs (A and B^T row stride)
#define LDA_ 40       // padded LDS row stride (bf16): conflict-free 16B loads

// GEMM1 fused: R = softmax_d((TE@Wq + bq) * Ksum/8) * Vsum  -> bf16
__global__ __launch_bounds__(256) void gemm1_softmax_kernel(
    const __bf16* __restrict__ A,    // TE bf16 [4096][1024]
    const __bf16* __restrict__ BT,   // Wq^T bf16 [1024][1024]
    const float* __restrict__ bq,
    const float* __restrict__ ksc,   // Ksum*0.125 [1024]
    const float* __restrict__ vsh,   // Vsum       [1024]
    __bf16* __restrict__ R)          // [4096][1024] bf16
{
  __shared__ __bf16 As[2][BM * LDA_];   // 2 x 20KB
  __shared__ __bf16 Bs[2][BN * LDA_];   // 2 x 5KB
  const int tid = threadIdx.x;
  const int wid = tid >> 5, lane = tid & 31;
  const int lane16 = lane & 15, laneHi = lane >> 4;
  const int n0 = blockIdx.x * BN;
  const int m0 = blockIdx.y * BM;

  v8f acc[8] = {};

  const int arow = tid;                               // one 32-elem A row/thread
  const int brow = tid >> 2, bseg = (tid & 3) * 8;    // 8 B^T elems/thread

  // LDS byte addresses (generic-pointer low 32 bits == LDS offset)
  const unsigned asLds = (unsigned)(uintptr_t)&As[0][0] + (unsigned)arow * (LDA_ * 2);
  const unsigned bsLds = (unsigned)(uintptr_t)&Bs[0][0] + (unsigned)(brow * LDA_ + bseg) * 2;
  const unsigned long long aG = (unsigned long long)(uintptr_t)A;
  const unsigned long long bG = (unsigned long long)(uintptr_t)BT;
  const unsigned aRowOff = (unsigned)((size_t)(m0 + arow) * KD * 2);
  const unsigned bRowOff = (unsigned)(((size_t)(n0 + brow) * KD + bseg) * 2);

  auto stage = [&](int k0, int buf) {
    const unsigned al = asLds + (unsigned)buf * (BM * LDA_ * 2);
    const unsigned ag = aRowOff + (unsigned)k0 * 2;
#pragma unroll
    for (int i = 0; i < 4; ++i) async_g2l_b128(al + 16 * i, ag + 16 * i, aG);
    async_g2l_b128(bsLds + (unsigned)buf * (BN * LDA_ * 2), bRowOff + (unsigned)k0 * 2, bG);
  };

  stage(0, 0);
  wait_async0();
  __syncthreads();
  int buf = 0;
  for (int k0 = 0; k0 < KD; k0 += BK) {
    const bool more = (k0 + BK) < KD;
    if (more) stage(k0 + BK, buf ^ 1);    // overlaps with WMMAs below
    Frag32 af[2];
#pragma unroll
    for (int r = 0; r < 2; ++r) {
      const uint4* ap = (const uint4*)(As[buf] + (wid * 32 + r * 16 + lane16) * LDA_ + laneHi * 8);
      af[r].u[0] = ap[0]; af[r].u[1] = ap[2];
    }
#pragma unroll
    for (int t = 0; t < 4; ++t) {
      Frag32 bfg;
      const uint4* bp = (const uint4*)(Bs[buf] + (t * 16 + lane16) * LDA_ + laneHi * 8);
      bfg.u[0] = bp[0]; bfg.u[1] = bp[2];
      acc[t]     = __builtin_amdgcn_wmma_f32_16x16x32_bf16(
          false, af[0].v, false, bfg.v, (short)0, acc[t], false, false);
      acc[4 + t] = __builtin_amdgcn_wmma_f32_16x16x32_bf16(
          false, af[1].v, false, bfg.v, (short)0, acc[4 + t], false, false);
    }
    if (more) { wait_async0(); __syncthreads(); buf ^= 1; }
  }

  // Epilogue: +bq, *Ksum/8, softmax over the 64 head columns, *Vsum, bf16 out.
  float kst[4], vvt[4], bqt[4];
#pragma unroll
  for (int t = 0; t < 4; ++t) {
    const int c = n0 + t * 16 + lane16;
    kst[t] = ksc[c]; vvt[t] = vsh[c]; bqt[t] = bq[c];
  }
#pragma unroll
  for (int r = 0; r < 2; ++r) {
    const int rbase = m0 + wid * 32 + r * 16 + laneHi * 8;
#pragma unroll
    for (int j = 0; j < 8; ++j) {
      float s0 = (acc[r * 4 + 0][j] + bqt[0]) * kst[0];
      float s1 = (acc[r * 4 + 1][j] + bqt[1]) * kst[1];
      float s2 = (acc[r * 4 + 2][j] + bqt[2]) * kst[2];
      float s3 = (acc[r * 4 + 3][j] + bqt[3]) * kst[3];
      float m = fmaxf(fmaxf(s0, s1), fmaxf(s2, s3));
      m = fmaxf(m, __shfl_xor(m, 1, 32));
      m = fmaxf(m, __shfl_xor(m, 2, 32));
      m = fmaxf(m, __shfl_xor(m, 4, 32));
      m = fmaxf(m, __shfl_xor(m, 8, 32));     // reduce within 16-lane half
      float e0 = __expf(s0 - m), e1 = __expf(s1 - m);
      float e2 = __expf(s2 - m), e3 = __expf(s3 - m);
      float sum = e0 + e1 + e2 + e3;
      sum += __shfl_xor(sum, 1, 32);
      sum += __shfl_xor(sum, 2, 32);
      sum += __shfl_xor(sum, 4, 32);
      sum += __shfl_xor(sum, 8, 32);
      const float inv = 1.0f / sum;
      __bf16* rp = R + (size_t)(rbase + j) * KD + n0 + lane16;
      rp[0]  = f2bf(e0 * inv * vvt[0]);
      rp[16] = f2bf(e1 * inv * vvt[1]);
      rp[32] = f2bf(e2 * inv * vvt[2]);
      rp[48] = f2bf(e3 * inv * vvt[3]);
    }
  }
}

// GEMM2: out = R @ Wo + bo   (f32 output [4096][4096])
__global__ __launch_bounds__(256) void gemm2_kernel(
    const __bf16* __restrict__ A,    // R bf16 [4096][1024]
    const __bf16* __restrict__ BT,   // Wo^T bf16 [4096][1024]
    const float* __restrict__ bo,
    float* __restrict__ out)
{
  __shared__ __bf16 As[2][BM * LDA_];
  __shared__ __bf16 Bs[2][BN * LDA_];
  const int tid = threadIdx.x;
  const int wid = tid >> 5, lane = tid & 31;
  const int lane16 = lane & 15, laneHi = lane >> 4;
  const int n0 = blockIdx.x * BN;
  const int m0 = blockIdx.y * BM;

  v8f acc[8] = {};

  const int arow = tid;
  const int brow = tid >> 2, bseg = (tid & 3) * 8;

  const unsigned asLds = (unsigned)(uintptr_t)&As[0][0] + (unsigned)arow * (LDA_ * 2);
  const unsigned bsLds = (unsigned)(uintptr_t)&Bs[0][0] + (unsigned)(brow * LDA_ + bseg) * 2;
  const unsigned long long aG = (unsigned long long)(uintptr_t)A;
  const unsigned long long bG = (unsigned long long)(uintptr_t)BT;
  const unsigned aRowOff = (unsigned)((size_t)(m0 + arow) * KD * 2);
  const unsigned bRowOff = (unsigned)(((size_t)(n0 + brow) * KD + bseg) * 2);

  auto stage = [&](int k0, int buf) {
    const unsigned al = asLds + (unsigned)buf * (BM * LDA_ * 2);
    const unsigned ag = aRowOff + (unsigned)k0 * 2;
#pragma unroll
    for (int i = 0; i < 4; ++i) async_g2l_b128(al + 16 * i, ag + 16 * i, aG);
    async_g2l_b128(bsLds + (unsigned)buf * (BN * LDA_ * 2), bRowOff + (unsigned)k0 * 2, bG);
  };

  stage(0, 0);
  wait_async0();
  __syncthreads();
  int buf = 0;
  for (int k0 = 0; k0 < KD; k0 += BK) {
    const bool more = (k0 + BK) < KD;
    if (more) stage(k0 + BK, buf ^ 1);
    Frag32 af[2];
#pragma unroll
    for (int r = 0; r < 2; ++r) {
      const uint4* ap = (const uint4*)(As[buf] + (wid * 32 + r * 16 + lane16) * LDA_ + laneHi * 8);
      af[r].u[0] = ap[0]; af[r].u[1] = ap[2];
    }
#pragma unroll
    for (int t = 0; t < 4; ++t) {
      Frag32 bfg;
      const uint4* bp = (const uint4*)(Bs[buf] + (t * 16 + lane16) * LDA_ + laneHi * 8);
      bfg.u[0] = bp[0]; bfg.u[1] = bp[2];
      acc[t]     = __builtin_amdgcn_wmma_f32_16x16x32_bf16(
          false, af[0].v, false, bfg.v, (short)0, acc[t], false, false);
      acc[4 + t] = __builtin_amdgcn_wmma_f32_16x16x32_bf16(
          false, af[1].v, false, bfg.v, (short)0, acc[4 + t], false, false);
    }
    if (more) { wait_async0(); __syncthreads(); buf ^= 1; }
  }

#pragma unroll
  for (int t = 0; t < 4; ++t) {
    const int c = n0 + t * 16 + lane16;
    const float bias = bo[c];
#pragma unroll
    for (int r = 0; r < 2; ++r) {
      const int rbase = m0 + wid * 32 + r * 16 + laneHi * 8;
#pragma unroll
      for (int j = 0; j < 8; ++j)
        out[(size_t)(rbase + j) * 4096 + c] = acc[r * 4 + t][j] + bias;
    }
  }
}

extern "C" void kernel_launch(void* const* d_in, const int* in_sizes, int n_in,
                              void* d_out, int out_size, void* d_ws, size_t ws_size,
                              hipStream_t stream) {
  (void)in_sizes; (void)n_in; (void)out_size; (void)ws_size;
  const float* TE = (const float*)d_in[0];
  const float* SE = (const float*)d_in[1];
  const float* VE = (const float*)d_in[2];
  const float* Wq = (const float*)d_in[3];
  const float* bq = (const float*)d_in[4];
  const float* Wk = (const float*)d_in[5];
  const float* bk = (const float*)d_in[6];
  const float* Wv = (const float*)d_in[7];
  const float* bv = (const float*)d_in[8];
  const float* Wo = (const float*)d_in[9];
  const float* bo = (const float*)d_in[10];
  float* out = (float*)d_out;

  char* ws = (char*)d_ws;
  size_t off = 0;
  auto take = [&](size_t bytes) {
    char* p = ws + off;
    off = (off + bytes + 255) & ~(size_t)255;
    return p;
  };
  float*  ssum = (float*)take(4096 * 4);
  float*  vsum = (float*)take(4096 * 4);
  float*  ksc  = (float*)take(1024 * 4);
  float*  vsh  = (float*)take(1024 * 4);
  __bf16* TEb  = (__bf16*)take((size_t)4096 * 1024 * 2);   // TE bf16
  __bf16* WqT  = (__bf16*)take((size_t)1024 * 1024 * 2);   // Wq^T bf16 [N][K]
  __bf16* WoT  = (__bf16*)take((size_t)4096 * 1024 * 2);   // Wo^T bf16 [N][K]
  __bf16* Rb   = (__bf16*)take((size_t)4096 * 1024 * 2);   // R bf16
  // total workspace ~27.3 MB

  colsum_kernel<<<dim3(16, 2), 256, 0, stream>>>(SE, VE, ssum, vsum);
  headsum_kernel<<<dim3(4, 2), 256, 0, stream>>>(ssum, vsum, Wk, bk, Wv, bv, ksc, vsh);
  cvt_bf16_kernel<<<4096, 256, 0, stream>>>(TE, TEb, 4096 * 1024);
  cvt_bf16_t_kernel<<<dim3(32, 32), 256, 0, stream>>>(Wq, WqT, 1024, 1024);
  cvt_bf16_t_kernel<<<dim3(128, 32), 256, 0, stream>>>(Wo, WoT, 4096, 1024);
  gemm1_softmax_kernel<<<dim3(16, 16), 256, 0, stream>>>(TEb, WqT, bq, ksc, vsh, Rb);
  gemm2_kernel<<<dim3(64, 16), 256, 0, stream>>>(Rb, WoT, bo, out);
}
